// DeformabelKPN_32461362823313
// MI455X (gfx1250) — matrix operand order repaired and checked by reading, not compiled, not run on hardware
//
#include <hip/hip_runtime.h>
#include <hip/hip_bf16.h>

// ---------------------------------------------------------------------------
// Deformable KPN for MI455X (gfx1250, wave32, WMMA).
// All 3x3 convs run as implicit GEMM on v_wmma_f32_16x16x32_f16.
// Activations: f16, channel-major [Cpad][H][W], Cpad multiple of 32.
// Weights pre-packed on device into the exact A-fragment lane layout.
// Conv kernel templated on CinP so the K loop fully unrolls and all
// A/B fragment addresses fold into immediate offsets.
// ---------------------------------------------------------------------------

typedef __attribute__((ext_vector_type(16))) _Float16 v16h;
typedef __attribute__((ext_vector_type(8)))  float    v8f;

#define H_ 512
#define W_ 512
#define HW_ (H_ * W_)
#define B_ 4

// ---------------------------------------------------------------------------
// Pack fp32 OIHW weights -> f16 A-fragments, layout apk[mt][kc][lane][16],
// k = tap*CinP + ci.  Contiguous K order (same as B fragment):
//   lane<16 : M=lane,   elems 0..15 = K0..15
//   lane>=16: M=lane-16, elems 0..15 = K16..31
// mode 0: identity channel map (ci >= CinRefDim -> 0 pad)
// mode 1: cat permutation for ww1 (cat = [feat 0-127 | frames 128-132 | samp 133-159],
//         reference order = [frames 0-4 | feat 5-132 | samp 133-159])
// ---------------------------------------------------------------------------
__global__ void pack_weights(const float* __restrict__ wref, _Float16* __restrict__ apk,
                             int Cout, int CinRefDim, int CinP, int KC, int mode, int total)
{
    int idx = blockIdx.x * 256 + threadIdx.x;
    if (idx >= total) return;
    int e    = idx & 15;
    int lane = (idx >> 4) & 31;
    int kc   = (idx >> 9) % KC;
    int mt   = idx / (KC << 9);
    int m    = mt * 16 + (lane & 15);
    int kl   = (lane & 16) + e;
    int k    = kc * 32 + kl;
    int tap  = k / CinP;
    int ci   = k - tap * CinP;
    float v = 0.f;
    if (m < Cout) {
        int cr = -1;
        if (mode == 0) { if (ci < CinRefDim) cr = ci; }
        else           { cr = (ci < 128) ? (5 + ci) : (ci < 133) ? (ci - 128) : ci; }
        if (cr >= 0) {
            int ky = tap / 3, kx = tap % 3;
            v = wref[(((size_t)m * CinRefDim + cr) * 3 + ky) * 3 + kx];
        }
    }
    apk[idx] = (_Float16)v;
}

// ---------------------------------------------------------------------------
// 3x3 SAME conv, implicit GEMM on WMMA.
// Block = 256 threads = 8 waves = (2 cout tiles) x (4 pixel tiles of 16):
// each block produces 32 couts x 64 pixels of one row y.
// LDS: halo tile [3 rows][68 x][CINP] f16, channel-innermost so each lane's
// B fragment (16 consecutive K for its column) is two ds_load_b128 with
// immediate offsets.  B layout (contiguous K, same order as packed A):
//   lane<16 : N=lane,    K=0..15 ; lane>=16: N=lane-16, K=16..31
// A fragments stream from packed global weights with a pointer marching by
// a constant 1024 B per K-chunk -> base + imm24 addressing after unroll.
// ---------------------------------------------------------------------------
template <int CINP, bool RELU, bool OUTF32>
__global__ __launch_bounds__(256)
void conv3x3_wmma(const _Float16* __restrict__ in, int CpitchIn,
                  const _Float16* __restrict__ apk, const float* __restrict__ bias,
                  void* __restrict__ outv, int CoutReal, int CpitchOut, int nMT2)
{
    constexpr int CCH = CINP / 32;      // 32-channel chunks
    constexpr int KC  = 9 * CCH;        // total K chunks
    __shared__ _Float16 lds[3 * 68 * CINP];   // <= 65,280 B (CINP=160)

    const int x0  = blockIdx.x * 64;
    const int y   = blockIdx.y;
    const int b   = blockIdx.z / nMT2;
    const int mt2 = blockIdx.z - b * nMT2;
    const _Float16* inb = in + (size_t)b * CpitchIn * HW_;

    // ---- stage halo tile (x-contiguous global reads, channel-innermost LDS)
    constexpr int nelems = 198 * CINP;  // 3 rows * 66 x * CINP
    for (int t = threadIdx.x; t < nelems; t += 256) {
        int slot = t % 66;
        int r    = (t / 66) % 3;
        int ci   = t / 198;
        int yy   = y + r - 1;
        int xx   = x0 - 1 + slot;
        _Float16 v = (_Float16)0.f;
        if ((unsigned)yy < (unsigned)H_ && (unsigned)xx < (unsigned)W_)
            v = inb[(size_t)ci * HW_ + (size_t)yy * W_ + xx];
        lds[(r * 68 + slot) * CINP + ci] = v;
    }
    __syncthreads();

    const int lane = threadIdx.x & 31;
    const int wid  = threadIdx.x >> 5;
    const int nx0  = (wid & 3) << 4;            // pixel subtile base (0..48)
    const int mt   = mt2 * 2 + (wid >> 2);      // 16-cout tile index
    const int col  = lane & 15;
    const int krb  = lane & 16;                 // K base for B fragment

    // marching A pointer: +512 halves (1024 B) per K chunk
    const _Float16* ap = apk + ((size_t)mt * KC * 32 + lane) * 16;
    // per-lane B base; all tap/chunk offsets are compile-time immediates
    const _Float16* bbase = lds + (nx0 + col) * CINP + krb;

    v8f acc = {0.f, 0.f, 0.f, 0.f, 0.f, 0.f, 0.f, 0.f};
#pragma unroll
    for (int tap = 0; tap < 9; ++tap) {
        const int dy = tap / 3 - 1;
        const int dx = tap % 3 - 1;
        const int boff = ((dy + 1) * 68 + (dx + 1)) * CINP;   // constant
#pragma unroll
        for (int cc = 0; cc < CCH; ++cc) {
            v16h a  = *(const v16h*)ap;
            v16h bf = *(const v16h*)(bbase + boff + cc * 32);
            acc = __builtin_amdgcn_wmma_f32_16x16x32_f16(
                false, a, false, bf, (short)0, acc, false, false);
            ap += 512;
        }
    }

    // ---- epilogue: D layout (VGPR r: lane<16 -> M=r, lane>=16 -> M=r+8)
    const int x   = x0 + nx0 + col;
    const int mrb = krb >> 1;                   // 0 or 8
    const size_t pbase = (size_t)b * CpitchOut * HW_ + (size_t)y * W_ + x;
#pragma unroll
    for (int r = 0; r < 8; ++r) {
        int mg = mt * 16 + mrb + r;
        if (mg < CoutReal) {
            float v = acc[r] + bias[mg];
            if (RELU) v = fmaxf(v, 0.f);
            if (OUTF32) ((float*)outv)[pbase + (size_t)mg * HW_] = v;
            else ((_Float16*)outv)[pbase + (size_t)mg * HW_] = (_Float16)v;
        }
    }
}

// ---------------------------------------------------------------------------
// frames fp32 -> f16 padded buffer [B][32][H][W] (ch 5..31 zero)
// ---------------------------------------------------------------------------
__global__ void prep_frames(const float* __restrict__ frames, _Float16* __restrict__ fbf, int total)
{
    int idx = blockIdx.x * 256 + threadIdx.x;
    if (idx >= total) return;
    int p  = idx % HW_;
    int ci = (idx / HW_) % 32;
    int b  = idx / (32 * HW_);
    _Float16 v = (_Float16)0.f;
    if (ci < 5) v = (_Float16)frames[((size_t)b * 5 + ci) * HW_ + p];
    fbf[idx] = v;
}

// frames -> cat channels 128..132 (f16)
__global__ void cat_frames(const float* __restrict__ frames, _Float16* __restrict__ cat, int total)
{
    int idx = blockIdx.x * 256 + threadIdx.x;
    if (idx >= total) return;
    int p  = idx % HW_;
    int ci = (idx / HW_) % 5;
    int b  = idx / (5 * HW_);
    cat[((size_t)b * 160 + 128 + ci) * HW_ + p] = (_Float16)frames[idx];
}

// ---------------------------------------------------------------------------
// Trilinear sampler: fp32 frames + fp32 offsets -> samp fp32 [B][27][HW]
// and cat channels 133..159 (f16).  blockIdx.y = sample s, blockIdx.z = batch.
// ---------------------------------------------------------------------------
__global__ void sampler(const float* __restrict__ frames, const float* __restrict__ off,
                        float* __restrict__ samp, _Float16* __restrict__ cat)
{
    int p = blockIdx.x * 256 + threadIdx.x;
    if (p >= HW_) return;
    int s = blockIdx.y, b = blockIdx.z;
    int y = p / W_, x = p % W_;
    const float* ob = off + ((size_t)b * 81 + s * 3) * HW_ + p;
    float tc = ob[0]             + (float)(s / 9 + 1);          // + bt (= grid_t + 2)
    float yc = ob[(size_t)HW_]   + (float)((s / 3) % 3 - 1) + (float)y;
    float xc = ob[2*(size_t)HW_] + (float)(s % 3 - 1)       + (float)x;
    tc = fminf(fmaxf(tc, 0.f), 4.f);
    yc = fminf(fmaxf(yc, 0.f), (float)(H_ - 1));
    xc = fminf(fmaxf(xc, 0.f), (float)(W_ - 1));
    float t0 = floorf(tc), y0 = floorf(yc), x0 = floorf(xc);
    float wt = tc - t0, wy = yc - y0, wx = xc - x0;
    int t0i = (int)t0, y0i = (int)y0, x0i = (int)x0;
    int t1i = min(t0i + 1, 4), y1i = min(y0i + 1, H_ - 1), x1i = min(x0i + 1, W_ - 1);
    const float* fb = frames + (size_t)b * 5 * HW_;
    auto g = [&](int ti, int yi, int xi) -> float {
        return fb[((size_t)ti * H_ + yi) * W_ + xi];
    };
    float c00 = g(t0i, y0i, x0i) * (1.f - wx) + g(t0i, y0i, x1i) * wx;
    float c01 = g(t0i, y1i, x0i) * (1.f - wx) + g(t0i, y1i, x1i) * wx;
    float c10 = g(t1i, y0i, x0i) * (1.f - wx) + g(t1i, y0i, x1i) * wx;
    float c11 = g(t1i, y1i, x0i) * (1.f - wx) + g(t1i, y1i, x1i) * wx;
    float c0 = c00 * (1.f - wy) + c01 * wy;
    float c1 = c10 * (1.f - wy) + c11 * wy;
    float r  = c0 * (1.f - wt) + c1 * wt;
    samp[((size_t)b * 27 + s) * HW_ + p] = r;
    cat[((size_t)b * 160 + 133 + s) * HW_ + p] = (_Float16)r;
}

// ---------------------------------------------------------------------------
// Grouped weighted sum: gp[g] = 4 * sum_{j<6} samp[g*6+j]*wgt[g*6+j];
// avg = sum_g gp / 27.  d_out = [gp_pred 4x4xHW][avg 4x1xHW].
// ---------------------------------------------------------------------------
__global__ void final_reduce(const float* __restrict__ samp, const _Float16* __restrict__ wgt,
                             float* __restrict__ out)
{
    int p = blockIdx.x * 256 + threadIdx.x;
    if (p >= HW_) return;
    int b = blockIdx.y;
    const float*    sb = samp + (size_t)b * 27 * HW_ + p;
    const _Float16* wb = wgt  + (size_t)b * 32 * HW_ + p;
    float tot = 0.f;
#pragma unroll
    for (int g = 0; g < 4; ++g) {
        float acc = 0.f;
#pragma unroll
        for (int j = 0; j < 6; ++j) {
            int c = g * 6 + j;
            acc += sb[(size_t)c * HW_] * (float)wb[(size_t)c * HW_];
        }
        float gp = 4.f * acc;
        out[((size_t)b * 4 + g) * HW_ + p] = gp;
        tot += gp;
    }
    out[(size_t)16 * HW_ + (size_t)b * HW_ + p] = tot / 27.f;
}

// ---------------------------------------------------------------------------
template <int CINP>
static void launch_conv(const _Float16* in, int CpitchIn, const _Float16* apk,
                        const float* bias, void* out, int CoutReal, int CpitchOut,
                        int nMT2, bool relu, hipStream_t stream)
{
    dim3 grid(W_ / 64, H_, B_ * nMT2), block(256);
    if (relu)
        conv3x3_wmma<CINP, true, false><<<grid, block, 0, stream>>>(
            in, CpitchIn, apk, bias, out, CoutReal, CpitchOut, nMT2);
    else
        conv3x3_wmma<CINP, false, true><<<grid, block, 0, stream>>>(
            in, CpitchIn, apk, bias, out, CoutReal, CpitchOut, nMT2);
}

extern "C" void kernel_launch(void* const* d_in, const int* in_sizes, int n_in,
                              void* d_out, int out_size, void* d_ws, size_t ws_size,
                              hipStream_t stream)
{
    (void)in_sizes; (void)n_in; (void)out_size; (void)ws_size;
    const float* frames = (const float*)d_in[0];
    const float* ow1 = (const float*)d_in[1];  const float* ob1 = (const float*)d_in[2];
    const float* ow2 = (const float*)d_in[3];  const float* ob2 = (const float*)d_in[4];
    const float* ow3 = (const float*)d_in[5];  const float* ob3 = (const float*)d_in[6];
    const float* ww1 = (const float*)d_in[7];  const float* wb1 = (const float*)d_in[8];
    const float* ww2 = (const float*)d_in[9];  const float* wb2 = (const float*)d_in[10];
    const float* ww3 = (const float*)d_in[11]; const float* wb3 = (const float*)d_in[12];

    char* ws = (char*)d_ws;
    size_t cur = 0;
    auto alloc = [&](size_t bytes) -> char* {
        char* p = ws + cur;
        cur += (bytes + 255) & ~(size_t)255;
        return p;
    };

    // packed weights: apk[mt][kc][32][16] f16
    _Float16* a1 = (_Float16*)alloc((size_t)4 * 9  * 512 * 2);  // 64x(5->32pad)
    _Float16* a2 = (_Float16*)alloc((size_t)8 * 18 * 512 * 2);  // 128x64
    _Float16* a3 = (_Float16*)alloc((size_t)6 * 36 * 512 * 2);  // 81(->96)x128
    _Float16* a4 = (_Float16*)alloc((size_t)4 * 45 * 512 * 2);  // 64x160 (cat perm)
    _Float16* a5 = (_Float16*)alloc((size_t)4 * 18 * 512 * 2);  // 64x64
    _Float16* a6 = (_Float16*)alloc((size_t)2 * 18 * 512 * 2);  // 27(->32)x64

    // activation buffers (f16 channel-major); w1 aliases f1, w2 aliases off
    _Float16* fbf = (_Float16*)alloc((size_t)B_ * 32  * HW_ * 2);
    _Float16* f1  = (_Float16*)alloc((size_t)B_ * 64  * HW_ * 2);
    _Float16* cat = (_Float16*)alloc((size_t)B_ * 160 * HW_ * 2);
    float*    off = (float*)   alloc((size_t)B_ * 81  * HW_ * 4);
    _Float16* wgt = (_Float16*)alloc((size_t)B_ * 32  * HW_ * 2);
    float*    smp = (float*)   alloc((size_t)B_ * 27  * HW_ * 4);
    _Float16* w1  = f1;                 // f1 dead after conv2
    _Float16* w2  = (_Float16*)off;     // off dead after sampler

    auto pack = [&](const float* wref, _Float16* apk, int Cout, int CinRef,
                    int CinP, int KC, int MT, int mode) {
        int total = MT * KC * 512;
        pack_weights<<<(total + 255) / 256, 256, 0, stream>>>(
            wref, apk, Cout, CinRef, CinP, KC, mode, total);
    };
    pack(ow1, a1, 64,  5,   32,  9,  4, 0);
    pack(ow2, a2, 128, 64,  64,  18, 8, 0);
    pack(ow3, a3, 81,  128, 128, 36, 6, 0);
    pack(ww1, a4, 64,  160, 160, 45, 4, 1);
    pack(ww2, a5, 64,  64,  64,  18, 4, 0);
    pack(ww3, a6, 27,  64,  64,  18, 2, 0);

    {
        int total = B_ * 32 * HW_;
        prep_frames<<<(total + 255) / 256, 256, 0, stream>>>(frames, fbf, total);
    }

    // OffsetNet
    launch_conv<32> (fbf, 32,  a1, ob1, f1,  64,  64,  2, true,  stream);
    launch_conv<64> (f1,  64,  a2, ob2, cat, 128, 160, 4, true,  stream);  // feat -> cat[0:128]
    launch_conv<128>(cat, 160, a3, ob3, off, 81,  81,  3, false, stream);  // fp32 offsets

    {   // frames -> cat[128:133]
        int total = B_ * 5 * HW_;
        cat_frames<<<(total + 255) / 256, 256, 0, stream>>>(frames, cat, total);
    }
    {   // trilinear samples -> samp fp32 + cat[133:160]
        dim3 grid((HW_ + 255) / 256, 27, B_), block(256);
        sampler<<<grid, block, 0, stream>>>(frames, off, smp, cat);
    }

    // WeightsNet
    launch_conv<160>(cat, 160, a4, wb1, w1,  64, 64, 2, true, stream);
    launch_conv<64> (w1,  64,  a5, wb2, w2,  64, 64, 2, true, stream);
    launch_conv<64> (w2,  64,  a6, wb3, wgt, 27, 32, 1, true, stream);

    {   // grouped weighted sum + average
        dim3 grid((HW_ + 255) / 256, B_), block(256);
        final_reduce<<<grid, block, 0, stream>>>(smp, wgt, (float*)d_out);
    }
}
